// SeparableFiberBundleConvFC_44049184587981
// MI455X (gfx1250) — matrix operand order repaired and checked by reading, not compiled, not run on hardware
//
#include <hip/hip_runtime.h>
#include <hip/hip_bf16.h>

// B=4, N=256, M=256, O=16, P=16, CIN=32, COUT=32, KDIM=16
// Step1+2 fused:  x1[b,m,o,c] = sum_{n,k} KB[b,m,n,o,k] * x[b,n,o,c] * Wk[c,k]
//   per (b,o): GEMM M=256 x K=4096(nk) x N=32(c), A=KB (f32->bf16 on the fly),
//   K split 4 ways across waves -> f32 partials, then reduce+convert.
// Step3: fkB[pd][oc] = sum_k fkb[p,o,k]*Wf[d*32+c,k]   (bf16, 512x512)
// Step4: out[(b,m)][pd] = (1/1024) * sum_oc x1bf[(b,m)][oc]*fkB[pd][oc] + bias[d]

typedef __attribute__((ext_vector_type(16))) __bf16 bf16x16;
typedef __attribute__((ext_vector_type(2)))  __bf16 bf16x2;
typedef __attribute__((ext_vector_type(8)))  float  f32x8;

union Op16 {
    bf16x16 v;
    unsigned u[8];
    bf16x2  h[8];
    uint4   q[2];
};

union BPair {
    unsigned u;
    bf16x2   h;
};

// round-half-up f32 -> bf16 pair, packed into one dword via v_perm_b32
__device__ __forceinline__ unsigned pack_bf2(float lo, float hi) {
    unsigned a = __float_as_uint(lo) + 0x8000u;
    unsigned b = __float_as_uint(hi) + 0x8000u;
    return __builtin_amdgcn_perm(b, a, 0x07060302u);   // D = {b[31:16], a[31:16]}
}

// round-half-up f32 -> bf16, splat into both halves of a dword
__device__ __forceinline__ bf16x2 splat_bf(float f) {
    unsigned t = __float_as_uint(f) + 0x8000u;
    BPair r;
    r.u = __builtin_amdgcn_perm(t, t, 0x07060302u);
    return r.h;
}

__device__ __forceinline__ unsigned short f2bf(float f) {
    unsigned a = __float_as_uint(f) + 0x8000u;
    return (unsigned short)(a >> 16);
}

// ---------------------------------------------------------------------------
// Kernel A: fused spatial aggregation with 4-way K split.
// 4096 wave tasks = 64(b,o) * 16(m-tile) * 4(k-split); 512 blocks * 8 waves.
// Wave owns 16(m) x 32(c) f32 accumulators over 32 nk-chunks, depth-3 pipeline.
// ---------------------------------------------------------------------------
__global__ __launch_bounds__(256) void k_spatial(
    const float* __restrict__ x,         // [4,256,16,32]
    const float* __restrict__ kb,        // [4,256,256,16,16]
    const float* __restrict__ Wk,        // [32,16]
    float* __restrict__ x1p)             // [4][1024][512] f32 partials
{
    const int lane = threadIdx.x & 31;
    const int wave = threadIdx.x >> 5;
    const int r = lane & 15;       // row/col within 16-tile
    const int g = lane >> 4;       // lane half -> K sub-range

    const int task  = blockIdx.x * 8 + wave;   // 0..4095
    const int s     = task & 3;                // k-split slice
    const int mtile = (task >> 2) & 15;
    const int bo    = task >> 6;
    const int b = bo >> 4;
    const int o = bo & 15;
    const int mbase = mtile * 16;
    const int m = mbase + r;

    // Per-lane Wk rows pre-packed to bf16x2: ctile0 -> c=r, ctile1 -> c=16+r
    Op16 wkp0, wkp1;
#pragma unroll
    for (int i = 0; i < 4; ++i) {
        float4 t = ((const float4*)(Wk + r * 16))[i];
        wkp0.u[2*i+0] = pack_bf2(t.x, t.y);
        wkp0.u[2*i+1] = pack_bf2(t.z, t.w);
        float4 w = ((const float4*)(Wk + (16 + r) * 16))[i];
        wkp1.u[2*i+0] = pack_bf2(w.x, w.y);
        wkp1.u[2*i+1] = pack_bf2(w.z, w.w);
    }

    f32x8 acc0 = {0.f,0.f,0.f,0.f,0.f,0.f,0.f,0.f};
    f32x8 acc1 = {0.f,0.f,0.f,0.f,0.f,0.f,0.f,0.f};

    // KB element (b,m,n,o,k) at b*16777216 + m*65536 + n*256 + o*16 + k
    const float* kbp = kb + (size_t)b * 16777216 + (size_t)m * 65536
                          + (size_t)o * 16;
    const int xBase = b * 131072 + o * 32;            // x (b,n,o,c)
    const int jbeg = s * 32;

    // depth-3 rotating prefetch buffers
    float4 abuf[3][4];
    float  xbuf[3][2];
    auto loadChunk = [&](int j, int slot) {
        const float* p = kbp + (size_t)(2 * j + g) * 256;
        abuf[slot][0] = ((const float4*)p)[0];
        abuf[slot][1] = ((const float4*)p)[1];
        abuf[slot][2] = ((const float4*)p)[2];
        abuf[slot][3] = ((const float4*)p)[3];
        xbuf[slot][0] = x[xBase + (2 * j + g) * 512 + r];        // ctile0: c=r
        xbuf[slot][1] = x[xBase + (2 * j + g) * 512 + 16 + r];   // ctile1: c=16+r
    };

    loadChunk(jbeg + 0, 0);
    loadChunk(jbeg + 1, 1);

#pragma unroll
    for (int it = 0; it < 32; ++it) {
        const int slot  = it % 3;
        const int pslot = (it + 2) % 3;
        // prefetch 2 chunks ahead (wrap to jbeg: harmless in-bounds reload)
        const int jn = (it + 2 < 32) ? (jbeg + it + 2) : jbeg;
        loadChunk(jn, pslot);

        // A tile: rows m=r, kk = 16g..16g+15 (n = 2j+g, k = 0..15)
        Op16 A;
        A.u[0] = pack_bf2(abuf[slot][0].x, abuf[slot][0].y);
        A.u[1] = pack_bf2(abuf[slot][0].z, abuf[slot][0].w);
        A.u[2] = pack_bf2(abuf[slot][1].x, abuf[slot][1].y);
        A.u[3] = pack_bf2(abuf[slot][1].z, abuf[slot][1].w);
        A.u[4] = pack_bf2(abuf[slot][2].x, abuf[slot][2].y);
        A.u[5] = pack_bf2(abuf[slot][2].z, abuf[slot][2].w);
        A.u[6] = pack_bf2(abuf[slot][3].x, abuf[slot][3].y);
        A.u[7] = pack_bf2(abuf[slot][3].z, abuf[slot][3].w);

        // B[kk,c] = X[n(kk),c] * Wk[c,k(kk)]; lane half g holds kk 16g..16g+15
        bf16x2 xs0 = splat_bf(xbuf[slot][0]);
        bf16x2 xs1 = splat_bf(xbuf[slot][1]);
        Op16 B0, B1;
#pragma unroll
        for (int v = 0; v < 8; ++v) {
            B0.h[v] = xs0 * wkp0.h[v];
            B1.h[v] = xs1 * wkp1.h[v];
        }

        acc0 = __builtin_amdgcn_wmma_f32_16x16x32_bf16(false, A.v, false, B0.v,
                                                       (short)0, acc0, false, false);
        acc1 = __builtin_amdgcn_wmma_f32_16x16x32_bf16(false, A.v, false, B1.v,
                                                       (short)0, acc1, false, false);
    }

    // D layout: lane r = col, VGPR v -> row v + 8g.  f32 partial store.
    float* outp = x1p + (size_t)s * 524288;
#pragma unroll
    for (int v = 0; v < 8; ++v) {
        const int row = b * 256 + mbase + v + 8 * g;
        const size_t ob = (size_t)row * 512 + o * 32;
        outp[ob + r]      = acc0[v];
        outp[ob + 16 + r] = acc1[v];
    }
}

// ---------------------------------------------------------------------------
// Reduce the 4 K-split partials and convert to bf16.
// ---------------------------------------------------------------------------
__global__ __launch_bounds__(256) void k_reduce(
    const float* __restrict__ x1p,       // [4][1024][512]
    unsigned short* __restrict__ x1bf)   // [1024][512]
{
    const int idx = blockIdx.x * 256 + threadIdx.x;   // 0..524287
    float s = x1p[idx] + x1p[idx + 524288] + x1p[idx + 1048576] + x1p[idx + 1572864];
    x1bf[idx] = f2bf(s);
}

// ---------------------------------------------------------------------------
// Kernel B: build bf16 fiber-kernel matrix fkB[pd][oc]. 262144 threads, 16 FMA each.
// ---------------------------------------------------------------------------
__global__ __launch_bounds__(256) void k_fiber(
    const float* __restrict__ fkb,       // [16,16,16] (p,o,k)
    const float* __restrict__ Wf,        // [1024,16]  (f=d*32+c, k)
    unsigned short* __restrict__ fkB)    // [512][512] bf16 (pd, oc)
{
    const int idx = blockIdx.x * 256 + threadIdx.x;   // 0..262143
    const int pd = idx >> 9, oc = idx & 511;
    const int p = pd >> 5, d = pd & 31, o = oc >> 5, c = oc & 31;
    const float* fb = fkb + (p * 16 + o) * 16;
    const float* wf = Wf + (d * 32 + c) * 16;
    float s = 0.f;
#pragma unroll
    for (int k = 0; k < 16; ++k) s += fb[k] * wf[k];
    fkB[idx] = f2bf(s);
}

// ---------------------------------------------------------------------------
// Kernel C: out[(b,m)][pd] = x1bf @ fkB^T / 1024 + bias.
// GEMM [1024 x 512] * [512 x 512]^T, bf16 in, f32 accum, depth-2 pipeline.
// grid = 64 blocks * 8 waves; wave owns 16(m) x 64(pd), 16 k-chunks of 32.
// ---------------------------------------------------------------------------
__global__ __launch_bounds__(256) void k_out(
    const unsigned short* __restrict__ x1bf,  // [1024][512]
    const unsigned short* __restrict__ fkB,   // [512][512]
    const float* __restrict__ bias,           // [32]
    float* __restrict__ out)                  // [1024][512] f32
{
    const int lane = threadIdx.x & 31;
    const int wave = threadIdx.x >> 5;
    const int r = lane & 15;
    const int g = lane >> 4;
    const int task = blockIdx.x * 8 + wave;   // 0..511
    const int mtile = task >> 3;              // 0..63
    const int ngroup = task & 7;              // 0..7 (64 cols each)
    const int row = mtile * 16 + r;
    const int nbase = ngroup * 64;

    const unsigned short* aRow = x1bf + (size_t)row * 512 + 8 * g;
    const unsigned short* bRow = fkB + (size_t)(nbase + r) * 512 + 8 * g;

    f32x8 acc[4];
#pragma unroll
    for (int t = 0; t < 4; ++t) acc[t] = (f32x8){0.f,0.f,0.f,0.f,0.f,0.f,0.f,0.f};

    Op16 Ab[2];
    Op16 Bb[2][4];
    auto loadA = [&](int j, int sl) {
        const unsigned short* pa = aRow + j * 32;
        Ab[sl].q[0] = *(const uint4*)pa;
        Ab[sl].q[1] = *(const uint4*)(pa + 16);
    };
    auto loadB = [&](int j, int sl) {
#pragma unroll
        for (int t = 0; t < 4; ++t) {
            const unsigned short* pb = bRow + t * 8192 + j * 32;
            Bb[sl][t].q[0] = *(const uint4*)pb;
            Bb[sl][t].q[1] = *(const uint4*)(pb + 16);
        }
    };

    loadA(0, 0);
    loadB(0, 0);

#pragma unroll
    for (int j = 0; j < 16; ++j) {
        const int sl = j & 1, nsl = sl ^ 1;
        const int jn = (j + 1) & 15;    // wrap: harmless in-bounds reload
        loadA(jn, nsl);
        loadB(jn, nsl);
#pragma unroll
        for (int t = 0; t < 4; ++t)
            acc[t] = __builtin_amdgcn_wmma_f32_16x16x32_bf16(false, Ab[sl].v, false, Bb[sl][t].v,
                                                             (short)0, acc[t], false, false);
    }

#pragma unroll
    for (int t = 0; t < 4; ++t) {
        const int pd = nbase + t * 16 + r;
        const float bv = bias[pd & 31];
#pragma unroll
        for (int v = 0; v < 8; ++v) {
            const int rr = mtile * 16 + v + 8 * g;
            out[(size_t)rr * 512 + pd] = acc[t][v] * (1.0f / 1024.0f) + bv;
        }
    }
}

// ---------------------------------------------------------------------------
extern "C" void kernel_launch(void* const* d_in, const int* in_sizes, int n_in,
                              void* d_out, int out_size, void* d_ws, size_t ws_size,
                              hipStream_t stream) {
    const float* x    = (const float*)d_in[0];   // 524288
    const float* kb   = (const float*)d_in[1];   // 67108864
    const float* fkb  = (const float*)d_in[2];   // 4096
    const float* Wk   = (const float*)d_in[3];   // 512
    const float* Wf   = (const float*)d_in[4];   // 16384
    const float* bias = (const float*)d_in[5];   // 32
    float* out = (float*)d_out;                  // 524288 f32

    float*          x1p  = (float*)d_ws;                        // 4*1024*512*4 = 8 MB
    unsigned short* x1bf = (unsigned short*)(x1p + 4 * 524288); // 1 MB
    unsigned short* fkB  = x1bf + 1024 * 512;                   // 512 KB

    k_spatial<<<512, 256, 0, stream>>>(x, kb, Wk, x1p);
    k_reduce<<<2048, 256, 0, stream>>>(x1p, x1bf);
    k_fiber<<<1024, 256, 0, stream>>>(fkb, Wf, fkB);
    k_out<<<64, 256, 0, stream>>>(x1bf, fkB, bias, out);
}
